// UpConv_15625091022860
// MI455X (gfx1250) — matrix operand order repaired and checked by reading, not compiled, not run on hardware
//
#include <hip/hip_runtime.h>
#include <math.h>

#define BB   4
#define NPT  65536
#define CI   64
#define CO   32
#define RES  128
#define HWSZ (RES*RES)        /* 16384 */
#define HLF  64
#define NROW (BB*NPT)         /* 262144 */
#define NSEG (BB*HWSZ)        /* 65536  */
#define EPSB 1e-5f
#define PLN  (BB*CO*HWSZ)     /* 2097152 */

typedef __attribute__((ext_vector_type(16))) _Float16 v16h;
typedef __attribute__((ext_vector_type(8)))  float    v8f;

__device__ inline v8f wmma_f16(v16h a, v16h b, v8f c) {
  return __builtin_amdgcn_wmma_f32_16x16x32_f16(false, a, false, b, (short)0, c, false, false);
}

__device__ inline unsigned encf(float f) {
  unsigned u = __float_as_uint(f);
  return (u & 0x80000000u) ? ~u : (u | 0x80000000u);
}
__device__ inline float decf(unsigned u) {
  unsigned v = (u & 0x80000000u) ? (u & 0x7FFFFFFFu) : ~u;
  return __uint_as_float(v);
}

// ---------------------------------------------------------------------------
// Stage B: fused point MLP with WMMA.  Each wave handles 16 rows.
// ---------------------------------------------------------------------------
__global__ __launch_bounds__(256) void point_mlp_wmma(
    const float* __restrict__ cf,
    const float* __restrict__ Wk, const float* __restrict__ bk,
    const float* __restrict__ W1, const float* __restrict__ b1,
    const float* __restrict__ W2, const float* __restrict__ b2,
    const float* __restrict__ W3, const float* __restrict__ b3,
    float* __restrict__ k_out, float* __restrict__ v_out,
    float* __restrict__ c_out)
{
  __shared__ _Float16 sW[8192];   // Wk | W1 | W2 | W3 (each 2048)
  __shared__ _Float16 sX[8192];   // 128 rows x 64 feats
  __shared__ _Float16 sT[8192];   // per-wave 16x64 staging

  const int tid = threadIdx.x;
  for (int i = tid; i < 2048; i += 256) {
    sW[i]        = (_Float16)Wk[i];
    sW[2048 + i] = (_Float16)W1[i];
    sW[4096 + i] = (_Float16)W2[i];
    sW[6144 + i] = (_Float16)W3[i];
  }
  const int rowbase = blockIdx.x * 128;
  for (int i = tid; i < 8192; i += 256)
    sX[i] = (_Float16)cf[rowbase * 64 + i];
  __syncthreads();

  const int wv = tid >> 5, lane = tid & 31;
  const int l15 = lane & 15, hi = lane >> 4;
  _Float16* sTw = sT + wv * 1024;

  auto ldA = [&](const _Float16* X, int ld, int row0, int kb) {
    v16h a;
    int base = (row0 + l15) * ld + kb * 32 + hi * 8;
    #pragma unroll
    for (int i = 0; i < 8; i++) { a[i] = X[base + i]; a[8 + i] = X[base + 16 + i]; }
    return a;
  };
  auto ldB = [&](const _Float16* W, int ld, int nb, int kb) {
    v16h bf;
    int base = (nb * 16 + l15) * ld + kb * 32 + hi * 8;
    #pragma unroll
    for (int i = 0; i < 8; i++) { bf[i] = W[base + i]; bf[8 + i] = W[base + 16 + i]; }
    return bf;
  };

  const int row0 = rowbase + wv * 16;
  v16h A0 = ldA(sX, 64, wv * 16, 0);
  v16h A1 = ldA(sX, 64, wv * 16, 1);

  // ---- k = cf @ Wk.T + bk ----
  #pragma unroll
  for (int nb = 0; nb < 2; ++nb) {
    v8f acc = {};
    acc = wmma_f16(A0, ldB(sW, 64, nb, 0), acc);
    acc = wmma_f16(A1, ldB(sW, 64, nb, 1), acc);
    int col = nb * 16 + l15; float bias = bk[col];
    #pragma unroll
    for (int r = 0; r < 8; r++)
      k_out[(row0 + hi * 8 + r) * 32 + col] = acc[r] + bias;
  }

  // ---- v1 = relu(cf @ W1.T + b1) -> sTw ----
  #pragma unroll
  for (int nb = 0; nb < 2; ++nb) {
    v8f acc = {};
    acc = wmma_f16(A0, ldB(sW + 2048, 64, nb, 0), acc);
    acc = wmma_f16(A1, ldB(sW + 2048, 64, nb, 1), acc);
    int col = nb * 16 + l15; float bias = b1[col];
    #pragma unroll
    for (int r = 0; r < 8; r++)
      sTw[(r + 8 * hi) * 64 + col] = (_Float16)fmaxf(acc[r] + bias, 0.f);
  }
  __syncthreads();

  // ---- v2 = relu(v1 @ W2.T + b2) ----
  v16h Av1 = ldA(sTw, 64, 0, 0);
  v8f a2[4];
  #pragma unroll
  for (int nb = 0; nb < 4; ++nb) {
    v8f acc = {};
    a2[nb] = wmma_f16(Av1, ldB(sW + 4096, 32, nb, 0), acc);
  }
  __syncthreads();
  #pragma unroll
  for (int nb = 0; nb < 4; ++nb) {
    int col = nb * 16 + l15; float bias = b2[col];
    #pragma unroll
    for (int r = 0; r < 8; r++)
      sTw[(r + 8 * hi) * 64 + col] = (_Float16)fmaxf(a2[nb][r] + bias, 0.f);
  }
  __syncthreads();

  // ---- v = v2 @ W3.T + b3 ----
  v16h B0 = ldA(sTw, 64, 0, 0);
  v16h B1 = ldA(sTw, 64, 0, 1);
  #pragma unroll
  for (int nb = 0; nb < 2; ++nb) {
    v8f acc = {};
    acc = wmma_f16(B0, ldB(sW + 6144, 64, nb, 0), acc);
    acc = wmma_f16(B1, ldB(sW + 6144, 64, nb, 1), acc);
    int col = nb * 16 + l15; float bias = b3[col];
    #pragma unroll
    for (int r = 0; r < 8; r++) {
      int m = row0 + hi * 8 + r;
      float vv = acc[r] + bias;
      v_out[m * 32 + col] = vv;
      c_out[m * 32 + col] = vv;
    }
  }
}

// ---------------------------------------------------------------------------
// WMMA transpose-conv 2x2 stride 2 (64 -> 32). One wave = 16 input pixels,
// 4 parity taps reuse the same A tile (K = 64 = 2 chunks).
// ---------------------------------------------------------------------------
__global__ __launch_bounds__(256) void upconv_wmma(
    const float* __restrict__ up, const float* __restrict__ w,
    const float* __restrict__ bias, float* __restrict__ m)
{
  __shared__ _Float16 sP[128 * 64];       // [px][ci] transposed input
  __shared__ _Float16 sW[4 * 2048];       // [tap][o*64+ci]
  const int tid = threadIdx.x;
  const int b = blockIdx.x >> 5;
  const int pix0 = (blockIdx.x & 31) * 128;

  for (int i = tid; i < 8192; i += 256) {
    int px = i >> 6, ci = i & 63;
    sP[i] = (_Float16)up[((size_t)(b * CI + ci)) * (HLF * HLF) + pix0 + px];
  }
  for (int i = tid; i < 8192; i += 256) {
    int tap = i >> 11, rem = i & 2047;
    int o = rem >> 6, ci = rem & 63;
    sW[i] = (_Float16)w[((ci * CO + o) * 2 + (tap >> 1)) * 2 + (tap & 1)];
  }
  __syncthreads();

  const int wv = tid >> 5, lane = tid & 31;
  const int l15 = lane & 15, hi = lane >> 4;
  const int abase = (wv * 16 + l15) * 64 + hi * 8;

  v16h A0, A1;
  #pragma unroll
  for (int i = 0; i < 8; i++) {
    A0[i] = sP[abase + i];        A0[8 + i] = sP[abase + 16 + i];
    A1[i] = sP[abase + 32 + i];   A1[8 + i] = sP[abase + 48 + i];
  }

  #pragma unroll
  for (int tap = 0; tap < 4; ++tap) {
    const _Float16* Wt = sW + tap * 2048;
    int dh = tap >> 1, dw = tap & 1;
    #pragma unroll
    for (int nb = 0; nb < 2; ++nb) {
      v16h bf0, bf1;
      int n = nb * 16 + l15;
      #pragma unroll
      for (int i = 0; i < 8; i++) {
        bf0[i] = Wt[n * 64 + hi * 8 + i];      bf0[8 + i] = Wt[n * 64 + 16 + hi * 8 + i];
        bf1[i] = Wt[n * 64 + 32 + hi * 8 + i]; bf1[8 + i] = Wt[n * 64 + 48 + hi * 8 + i];
      }
      v8f acc = {};
      acc = wmma_f16(A0, bf0, acc);
      acc = wmma_f16(A1, bf1, acc);
      float bs = bias[n];
      #pragma unroll
      for (int r = 0; r < 8; r++) {
        int ip = pix0 + wv * 16 + r + 8 * hi;
        int h = ip >> 6, wc = ip & 63;
        m[((b * 64 + n) * HWSZ) + (2 * h + dh) * RES + (2 * wc + dw)] = acc[r] + bs;
      }
    }
  }
}

__global__ __launch_bounds__(256) void copy_down(
    const float* __restrict__ down, float* __restrict__ m)
{
  int i = blockIdx.x * 256 + threadIdx.x;   // PLN
  int pix = i & 16383, o = (i >> 14) & 31, b = i >> 19;
  m[((b * 64 + 32 + o) * HWSZ) + pix] = down[i];
}

// ---------------------------------------------------------------------------
// WMMA 3x3 conv, pad 1, 64 -> 32.  Tap-major im2col: k = tap*64+ci, so each
// 32-wide K chunk shares one (dy,dx).  Block = 8x16 spatial tile, wave = row.
// ---------------------------------------------------------------------------
__global__ __launch_bounds__(256) void conv3x3_wmma_64_32(
    const float* __restrict__ in, const float* __restrict__ w,
    const float* __restrict__ bias, float* __restrict__ out)
{
  __shared__ _Float16 sP[64 * 180];   // [ci][10 rows][18 cols] halo patch
  __shared__ _Float16 sWt[32 * 576];  // [n][k], k = tap*64+ci

  const int tid = threadIdx.x;
  const int b = blockIdx.z, y0 = blockIdx.y * 8, x0 = blockIdx.x * 16;

  for (int i = tid; i < 64 * 180; i += 256) {
    int ci = i / 180, rem = i - ci * 180;
    int r = rem / 18, c = rem - r * 18;
    int y = y0 + r - 1, x = x0 + c - 1;
    float v = (y >= 0 && y < RES && x >= 0 && x < RES)
                  ? in[((b * 64 + ci) * HWSZ) + y * RES + x] : 0.f;
    sP[i] = (_Float16)v;
  }
  for (int i = tid; i < 32 * 576; i += 256) {
    int n = i / 576, k = i - n * 576;
    int tap = k >> 6, ci = k & 63;
    sWt[i] = (_Float16)w[(n * 64 + ci) * 9 + tap];
  }
  __syncthreads();

  const int wv = tid >> 5, lane = tid & 31;
  const int l15 = lane & 15, hi = lane >> 4;

  v8f acc0 = {}, acc1 = {};
  #pragma unroll
  for (int kc = 0; kc < 18; ++kc) {
    const int tap = kc >> 1, ci0 = (kc & 1) << 5;
    const int dy = tap / 3, dx = tap - dy * 3;
    const int pbase = (wv + dy) * 18 + (l15 + dx);
    v16h a;
    #pragma unroll
    for (int i = 0; i < 8; i++) {
      a[i]     = sP[(ci0 + hi * 8 + i) * 180 + pbase];
      a[8 + i] = sP[(ci0 + 16 + hi * 8 + i) * 180 + pbase];
    }
    v16h b0, b1;
    const int kb = kc * 32 + hi * 8;
    #pragma unroll
    for (int i = 0; i < 8; i++) {
      b0[i] = sWt[l15 * 576 + kb + i];        b0[8 + i] = sWt[l15 * 576 + kb + 16 + i];
      b1[i] = sWt[(16 + l15) * 576 + kb + i]; b1[8 + i] = sWt[(16 + l15) * 576 + kb + 16 + i];
    }
    acc0 = wmma_f16(a, b0, acc0);
    acc1 = wmma_f16(a, b1, acc1);
  }
  const int y = y0 + wv;
  const float bs0 = bias[l15], bs1 = bias[16 + l15];
  #pragma unroll
  for (int r = 0; r < 8; r++) {
    int x = x0 + r + 8 * hi;
    out[((b * 32 + l15) * HWSZ) + y * RES + x]      = acc0[r] + bs0;
    out[((b * 32 + 16 + l15) * HWSZ) + y * RES + x] = acc1[r] + bs1;
  }
}

// ---------------------------------------------------------------------------
// WMMA 3x3 conv, pad 1, 32 -> 32, plus residual add: out = fp + conv(xup)+b
// ---------------------------------------------------------------------------
__global__ __launch_bounds__(256) void res_conv3x3_wmma(
    const float* __restrict__ fp, const float* __restrict__ xup,
    const float* __restrict__ w, const float* __restrict__ bias,
    float* __restrict__ out)
{
  __shared__ _Float16 sP[32 * 180];
  __shared__ _Float16 sWt[32 * 288];  // [n][k], k = tap*32+ci

  const int tid = threadIdx.x;
  const int b = blockIdx.z, y0 = blockIdx.y * 8, x0 = blockIdx.x * 16;

  for (int i = tid; i < 32 * 180; i += 256) {
    int ci = i / 180, rem = i - ci * 180;
    int r = rem / 18, c = rem - r * 18;
    int y = y0 + r - 1, x = x0 + c - 1;
    float v = (y >= 0 && y < RES && x >= 0 && x < RES)
                  ? xup[((b * 32 + ci) * HWSZ) + y * RES + x] : 0.f;
    sP[i] = (_Float16)v;
  }
  for (int i = tid; i < 32 * 288; i += 256) {
    int n = i / 288, k = i - n * 288;
    int tap = k >> 5, ci = k & 31;
    sWt[i] = (_Float16)w[(n * 32 + ci) * 9 + tap];
  }
  __syncthreads();

  const int wv = tid >> 5, lane = tid & 31;
  const int l15 = lane & 15, hi = lane >> 4;

  v8f acc0 = {}, acc1 = {};
  #pragma unroll
  for (int kc = 0; kc < 9; ++kc) {
    const int dy = kc / 3, dx = kc - dy * 3;
    const int pbase = (wv + dy) * 18 + (l15 + dx);
    v16h a;
    #pragma unroll
    for (int i = 0; i < 8; i++) {
      a[i]     = sP[(hi * 8 + i) * 180 + pbase];
      a[8 + i] = sP[(16 + hi * 8 + i) * 180 + pbase];
    }
    v16h b0, b1;
    const int kb = kc * 32 + hi * 8;
    #pragma unroll
    for (int i = 0; i < 8; i++) {
      b0[i] = sWt[l15 * 288 + kb + i];        b0[8 + i] = sWt[l15 * 288 + kb + 16 + i];
      b1[i] = sWt[(16 + l15) * 288 + kb + i]; b1[8 + i] = sWt[(16 + l15) * 288 + kb + 16 + i];
    }
    acc0 = wmma_f16(a, b0, acc0);
    acc1 = wmma_f16(a, b1, acc1);
  }
  const int y = y0 + wv;
  const float bs0 = bias[l15], bs1 = bias[16 + l15];
  #pragma unroll
  for (int r = 0; r < 8; r++) {
    int x = x0 + r + 8 * hi;
    size_t o0 = ((size_t)(b * 32 + l15) * HWSZ) + y * RES + x;
    size_t o1 = ((size_t)(b * 32 + 16 + l15) * HWSZ) + y * RES + x;
    out[o0] = fp[o0] + acc0[r] + bs0;
    out[o1] = fp[o1] + acc1[r] + bs1;
  }
}

// ---------------------------------------------------------------------------
// WMMA 1x1 conv 32 -> 32 (used for convq and convf): pure K=32 GEMM.
// ---------------------------------------------------------------------------
__global__ __launch_bounds__(256) void conv1x1_wmma(
    const float* __restrict__ in, const float* __restrict__ w,
    const float* __restrict__ bias, float* __restrict__ out)
{
  __shared__ _Float16 sP[128 * 32];   // [px][ci]
  __shared__ _Float16 sW[32 * 32];    // [n][k]
  const int tid = threadIdx.x;
  const int b = blockIdx.x >> 7;
  const int pix0 = (blockIdx.x & 127) * 128;

  for (int i = tid; i < 4096; i += 256) {
    int px = i >> 5, ci = i & 31;
    sP[i] = (_Float16)in[((size_t)(b * 32 + ci)) * HWSZ + pix0 + px];
  }
  for (int i = tid; i < 1024; i += 256)
    sW[i] = (_Float16)w[i];
  __syncthreads();

  const int wv = tid >> 5, lane = tid & 31;
  const int l15 = lane & 15, hi = lane >> 4;
  const int abase = (wv * 16 + l15) * 32 + hi * 8;

  v16h a;
  #pragma unroll
  for (int i = 0; i < 8; i++) { a[i] = sP[abase + i]; a[8 + i] = sP[abase + 16 + i]; }

  v16h b0, b1;
  #pragma unroll
  for (int i = 0; i < 8; i++) {
    b0[i] = sW[l15 * 32 + hi * 8 + i];        b0[8 + i] = sW[l15 * 32 + 16 + hi * 8 + i];
    b1[i] = sW[(16 + l15) * 32 + hi * 8 + i]; b1[8 + i] = sW[(16 + l15) * 32 + 16 + hi * 8 + i];
  }
  v8f acc0 = {}, acc1 = {};
  acc0 = wmma_f16(a, b0, acc0);
  acc1 = wmma_f16(a, b1, acc1);

  const float bs0 = bias[l15], bs1 = bias[16 + l15];
  #pragma unroll
  for (int r = 0; r < 8; r++) {
    int px = pix0 + wv * 16 + r + 8 * hi;
    out[((size_t)(b * 32 + l15)) * HWSZ + px]      = acc0[r] + bs0;
    out[((size_t)(b * 32 + 16 + l15)) * HWSZ + px] = acc1[r] + bs1;
  }
}

// ---------------------------------------------------------------------------
__global__ __launch_bounds__(256) void zero_seg(unsigned* __restrict__ smax,
                                                float* __restrict__ denom,
                                                float* __restrict__ num,
                                                float* __restrict__ stats)
{
  int i = blockIdx.x * 256 + threadIdx.x;
  if (i < NSEG * 4) { smax[i] = 0u; denom[i] = 0.f; }
  if (i < NSEG * 32) num[i] = 0.f;
  if (i < 256) stats[i] = 0.f;
}

// per-channel sum / sumsq over (B,H,W); blockIdx.y = channel
__global__ __launch_bounds__(256) void chan_stats(
    const float* __restrict__ in, float* __restrict__ stats, int C)
{
  int c = blockIdx.y;
  int i = blockIdx.x * 256 + threadIdx.x;   // over B*HW = 65536
  int b = i >> 14, pix = i & 16383;
  float v = in[((b * C + c) * HWSZ) + pix];
  __shared__ float s1[256], s2[256];
  int t = threadIdx.x;
  s1[t] = v; s2[t] = v * v; __syncthreads();
  for (int s = 128; s > 0; s >>= 1) {
    if (t < s) { s1[t] += s1[t + s]; s2[t] += s2[t + s]; }
    __syncthreads();
  }
  if (t == 0) { atomicAdd(&stats[2 * c], s1[0]); atomicAdd(&stats[2 * c + 1], s2[0]); }
}

__global__ __launch_bounds__(256) void bn_apply(
    const float* __restrict__ in, const float* __restrict__ stats,
    const float* __restrict__ g, const float* __restrict__ bb,
    float* __restrict__ out, float invcnt, int relu)
{
  int i = blockIdx.x * 256 + threadIdx.x;
  int c = (i >> 14) & 31;
  float mean = stats[2 * c] * invcnt;
  float var  = stats[2 * c + 1] * invcnt - mean * mean;
  float y = g[c] * (in[i] - mean) * rsqrtf(var + EPSB) + bb[c];
  out[i] = relu ? fmaxf(y, 0.f) : y;
}

__device__ inline void breduce(float v, float* target, float* red)
{
  int t = threadIdx.x;
  red[t] = v; __syncthreads();
  for (int s = 128; s > 0; s >>= 1) {
    if (t < s) red[t] += red[t + s];
    __syncthreads();
  }
  if (t == 0) atomicAdd(target, red[0]);
  __syncthreads();
}

// idx/pos + first pos-MLP layer + BN stats (2 ch over all points)
__global__ __launch_bounds__(256) void pos_stage1(
    const float* __restrict__ p, const float* __restrict__ w1,
    const float* __restrict__ b1, int* __restrict__ idxb,
    float* __restrict__ hpos, float* __restrict__ stats, int i0, int i1)
{
  int i = blockIdx.x * 256 + threadIdx.x;   // NROW
  float px = p[i * 3 + i0], py = p[i * 3 + i1];
  float fx = fminf(fmaxf(px / 1.101f + 0.5f, 0.f), 0.999f);
  float fy = fminf(fmaxf(py / 1.101f + 0.5f, 0.f), 0.999f);
  int cx = (int)(fx * 128.f), cy = (int)(fy * 128.f);
  idxb[i] = cx + 128 * cy;
  float qx = fx - floorf(fx * 128.f) * (1.f / 128.f);
  float qy = fy - floorf(fy * 128.f) * (1.f / 128.f);
  float h0 = qx * w1[0] + qy * w1[1] + b1[0];
  float h1 = qx * w1[2] + qy * w1[3] + b1[1];
  hpos[i * 2] = h0; hpos[i * 2 + 1] = h1;
  __shared__ float red[256];
  breduce(h0, &stats[0], red);
  breduce(h1, &stats[1], red);
  breduce(h0 * h0, &stats[2], red);
  breduce(h1 * h1, &stats[3], red);
}

// BN+relu on hpos, pos_enc = second linear, w = k - qg + pos_enc, wf = fcw(w)
__global__ __launch_bounds__(256) void pos_stage2(
    const float* __restrict__ hpos, const float* __restrict__ stats,
    const float* __restrict__ g, const float* __restrict__ bb,
    const float* __restrict__ w2, const float* __restrict__ b2,
    const float* __restrict__ kbuf, const float* __restrict__ qp,
    const int* __restrict__ idxb,
    const float* __restrict__ fww, const float* __restrict__ fwb,
    float* __restrict__ pos_enc, float* __restrict__ wfb,
    float* __restrict__ statsW)
{
  int i = blockIdx.x * 256 + threadIdx.x;   // NROW
  const float invc = 1.f / (float)NROW;
  float m0 = stats[0] * invc, m1 = stats[1] * invc;
  float v0 = stats[2] * invc - m0 * m0, v1 = stats[3] * invc - m1 * m1;
  float h0 = fmaxf(g[0] * (hpos[i * 2] - m0) * rsqrtf(v0 + EPSB) + bb[0], 0.f);
  float h1 = fmaxf(g[1] * (hpos[i * 2 + 1] - m1) * rsqrtf(v1 + EPSB) + bb[1], 0.f);
  int b = i >> 16;
  int idxv = idxb[i];
  const float* qb = qp + ((size_t)b * 32) * HWSZ + idxv;
  float wf[4] = {0.f, 0.f, 0.f, 0.f};
  #pragma unroll 8
  for (int j = 0; j < 32; ++j) {
    float pe = w2[j * 2] * h0 + w2[j * 2 + 1] * h1 + b2[j];
    pos_enc[(size_t)i * 32 + j] = pe;
    float wvv = kbuf[(size_t)i * 32 + j] - qb[j * HWSZ] + pe;
    wf[0] += fww[j] * wvv;
    wf[1] += fww[32 + j] * wvv;
    wf[2] += fww[64 + j] * wvv;
    wf[3] += fww[96 + j] * wvv;
  }
  #pragma unroll
  for (int t = 0; t < 4; ++t) { wf[t] += fwb[t]; wfb[i * 4 + t] = wf[t]; }
  __shared__ float red[256];
  for (int t = 0; t < 4; ++t) breduce(wf[t], &statsW[t], red);
  for (int t = 0; t < 4; ++t) breduce(wf[t] * wf[t], &statsW[4 + t], red);
}

// BN+relu on wf (in place) then segment atomic-max via ordered uints
__global__ __launch_bounds__(256) void seg_max_k(
    float* __restrict__ wfb, const float* __restrict__ statsW,
    const float* __restrict__ g, const float* __restrict__ bb,
    const int* __restrict__ idxb, unsigned* __restrict__ smax)
{
  int i = blockIdx.x * 256 + threadIdx.x;
  int seg = (i >> 16) * HWSZ + idxb[i];
  const float invc = 1.f / (float)NROW;
  #pragma unroll
  for (int t = 0; t < 4; ++t) {
    float mean = statsW[t] * invc;
    float var  = statsW[4 + t] * invc - mean * mean;
    float y = fmaxf(g[t] * (wfb[i * 4 + t] - mean) * rsqrtf(var + EPSB) + bb[t], 0.f);
    wfb[i * 4 + t] = y;
    atomicMax(&smax[seg * 4 + t], encf(y));
  }
}

// e = exp(wf - max); denom += e; num += (v + pos_enc) * e[j%4]
__global__ __launch_bounds__(256) void scatter_k(
    const float* __restrict__ wfb, const unsigned* __restrict__ smax,
    const float* __restrict__ vbuf, const float* __restrict__ pos_enc,
    const int* __restrict__ idxb, float* __restrict__ denom,
    float* __restrict__ num)
{
  int i = blockIdx.x * 256 + threadIdx.x;
  int seg = (i >> 16) * HWSZ + idxb[i];
  float e[4];
  #pragma unroll
  for (int t = 0; t < 4; ++t) {
    e[t] = expf(wfb[i * 4 + t] - decf(smax[seg * 4 + t]));
    atomicAdd(&denom[seg * 4 + t], e[t]);
  }
  #pragma unroll 8
  for (int j = 0; j < 32; ++j) {
    float val = vbuf[(size_t)i * 32 + j] + pos_enc[(size_t)i * 32 + j];
    atomicAdd(&num[(size_t)seg * 32 + j], val * e[j & 3]);
  }
}

// fea -> plane (B,32,128,128)
__global__ __launch_bounds__(256) void fea_k(
    const float* __restrict__ num, const float* __restrict__ denom,
    float* __restrict__ fp)
{
  int i = blockIdx.x * 256 + threadIdx.x;
  int pix = i & 16383, o = (i >> 14) & 31, b = i >> 19;
  int seg = b * HWSZ + pix;
  float d = denom[seg * 4 + (o & 3)];
  fp[i] = (d > 0.f) ? num[(size_t)seg * 32 + o] / d : 0.f;
}

// depthwise 3x3, pad 1, no bias
__global__ __launch_bounds__(256) void dwconv3x3(
    const float* __restrict__ in, const float* __restrict__ w,
    float* __restrict__ out)
{
  int i = blockIdx.x * 256 + threadIdx.x;
  int x = i & 127, y = (i >> 7) & 127, o = (i >> 14) & 31, b = i >> 19;
  const float* ib = in + ((b * 32 + o) * HWSZ);
  __builtin_prefetch(ib + y * RES + x + RES, 0, 1);
  float acc = 0.f;
  #pragma unroll
  for (int ky = 0; ky < 3; ++ky) {
    int yy = y + ky - 1;
    if (yy < 0 || yy > 127) continue;
    #pragma unroll
    for (int kx = 0; kx < 3; ++kx) {
      int xx = x + kx - 1;
      if (xx < 0 || xx > 127) continue;
      acc += ib[yy * RES + xx] * w[o * 9 + ky * 3 + kx];
    }
  }
  out[i] = acc;
}

// bilinear sample of final plane at points, accumulate into c (B,N,32)
__global__ __launch_bounds__(256) void sample_add(
    const float* __restrict__ plane, const float* __restrict__ p,
    float* __restrict__ cout, int i0, int i1)
{
  int i = blockIdx.x * 256 + threadIdx.x;   // NROW
  int b = i >> 16;
  float px = p[i * 3 + i0], py = p[i * 3 + i1];
  float fx = fminf(fmaxf(px / 1.101f + 0.5f, 0.f), 0.999f);
  float fy = fminf(fmaxf(py / 1.101f + 0.5f, 0.f), 0.999f);
  float ix = fminf(fx * 127.f, 127.f);
  float iy = fminf(fy * 127.f, 127.f);
  int x0 = (int)ix, y0 = (int)iy;
  float wx = ix - (float)x0, wy = iy - (float)y0;
  int x1 = min(x0 + 1, 127), y1 = min(y0 + 1, 127);
  float w00 = (1.f - wx) * (1.f - wy), w01 = wx * (1.f - wy);
  float w10 = (1.f - wx) * wy,         w11 = wx * wy;
  const float* pb = plane + (size_t)b * 32 * HWSZ;
  #pragma unroll 8
  for (int ch = 0; ch < 32; ++ch) {
    const float* pc = pb + ch * HWSZ;
    float val = pc[y0 * RES + x0] * w00 + pc[y0 * RES + x1] * w01 +
                pc[y1 * RES + x0] * w10 + pc[y1 * RES + x1] * w11;
    cout[(size_t)i * 32 + ch] += val;
  }
}

// ---------------------------------------------------------------------------
extern "C" void kernel_launch(void* const* d_in, const int* in_sizes, int n_in,
                              void* d_out, int out_size, void* d_ws, size_t ws_size,
                              hipStream_t stream)
{
  (void)in_sizes; (void)n_in; (void)out_size; (void)ws_size;
  const float* p        = (const float*)d_in[0];
  const float* c_last   = (const float*)d_in[1];
  const float* ups[3]   = {(const float*)d_in[2], (const float*)d_in[3], (const float*)d_in[4]};
  const float* downs[3] = {(const float*)d_in[5], (const float*)d_in[6], (const float*)d_in[7]};
  const float* upconv_w = (const float*)d_in[8];
  const float* upconv_b = (const float*)d_in[9];
  const float* fck_w  = (const float*)d_in[10]; const float* fck_b  = (const float*)d_in[11];
  const float* fcv_w1 = (const float*)d_in[12]; const float* fcv_b1 = (const float*)d_in[13];
  const float* fcv_w2 = (const float*)d_in[14]; const float* fcv_b2 = (const float*)d_in[15];
  const float* fcv_w3 = (const float*)d_in[16]; const float* fcv_b3 = (const float*)d_in[17];
  const float* convq_w = (const float*)d_in[18]; const float* convq_b = (const float*)d_in[19];
  const float* fcpos_w1 = (const float*)d_in[20]; const float* fcpos_b1 = (const float*)d_in[21];
  const float* fcpos_bn_g = (const float*)d_in[22]; const float* fcpos_bn_b = (const float*)d_in[23];
  const float* fcpos_w2 = (const float*)d_in[24]; const float* fcpos_b2 = (const float*)d_in[25];
  const float* fcw_w = (const float*)d_in[26]; const float* fcw_b = (const float*)d_in[27];
  const float* fcw_bn_g = (const float*)d_in[28]; const float* fcw_bn_b = (const float*)d_in[29];
  const float* conv1_w = (const float*)d_in[30]; const float* conv1_b = (const float*)d_in[31];
  const float* conv1_bn_g = (const float*)d_in[32]; const float* conv1_bn_b = (const float*)d_in[33];
  const float* wconv_w1 = (const float*)d_in[34]; const float* wconv_w2 = (const float*)d_in[35];
  const float* c1x1_w = (const float*)d_in[36]; const float* c1x1_b = (const float*)d_in[37];
  const float* convf_w = (const float*)d_in[38]; const float* convf_b = (const float*)d_in[39];
  const float* convf_bn_g = (const float*)d_in[40]; const float* convf_bn_b = (const float*)d_in[41];

  float* out = (float*)d_out;
  float* xout[3] = {out, out + PLN, out + 2 * PLN};
  float* cout = out + 3 * PLN;

  float* ws      = (float*)d_ws;
  float* k_buf   = ws;
  float* v_buf   = k_buf + (size_t)NROW * 32;
  float* pos_enc = v_buf + (size_t)NROW * 32;
  float* m_buf   = pos_enc + (size_t)NROW * 32;
  float* num     = m_buf;                    // alias: zeroed after conv1 consumes m_buf
  float* h_pre   = m_buf + (size_t)BB * 64 * HWSZ;
  float* x_pl    = h_pre + PLN;
  float* qp      = x_pl + PLN;
  float* fp_a    = qp + PLN;
  float* fp_b    = fp_a + PLN;
  float* wfb     = fp_b + PLN;
  float* hpos    = wfb + (size_t)NROW * 4;
  int*   idxb    = (int*)(hpos + (size_t)NROW * 2);
  unsigned* smax = (unsigned*)(idxb + NROW);
  float* denom   = (float*)(smax + (size_t)NSEG * 4);
  float* stats   = denom + (size_t)NSEG * 4;

  const int TB = 256;

  // Stage B: WMMA point MLPs (k, v; c initialized with v)
  point_mlp_wmma<<<NROW / 128, TB, 0, stream>>>(
      c_last, fck_w, fck_b, fcv_w1, fcv_b1, fcv_w2, fcv_b2, fcv_w3, fcv_b3,
      k_buf, v_buf, cout);

  const int IJ[3][2] = {{0, 1}, {0, 2}, {1, 2}};
  for (int pl = 0; pl < 3; ++pl) {
    upconv_wmma<<<BB * 32, TB, 0, stream>>>(ups[pl], upconv_w, upconv_b, m_buf);
    copy_down<<<PLN / TB, TB, 0, stream>>>(downs[pl], m_buf);
    conv3x3_wmma_64_32<<<dim3(8, 16, BB), TB, 0, stream>>>(m_buf, conv1_w, conv1_b, h_pre);
    zero_seg<<<PLN / TB, TB, 0, stream>>>(smax, denom, num, stats);
    chan_stats<<<dim3(256, 32), TB, 0, stream>>>(h_pre, stats + 0, 32);
    bn_apply<<<PLN / TB, TB, 0, stream>>>(h_pre, stats + 0, conv1_bn_g, conv1_bn_b,
                                          x_pl, 1.f / 65536.f, 1);
    conv1x1_wmma<<<BB * 128, TB, 0, stream>>>(x_pl, convq_w, convq_b, qp);
    pos_stage1<<<NROW / TB, TB, 0, stream>>>(p, fcpos_w1, fcpos_b1, idxb, hpos,
                                             stats + 64, IJ[pl][0], IJ[pl][1]);
    pos_stage2<<<NROW / TB, TB, 0, stream>>>(hpos, stats + 64, fcpos_bn_g, fcpos_bn_b,
                                             fcpos_w2, fcpos_b2, k_buf, qp, idxb,
                                             fcw_w, fcw_b, pos_enc, wfb, stats + 72);
    seg_max_k<<<NROW / TB, TB, 0, stream>>>(wfb, stats + 72, fcw_bn_g, fcw_bn_b, idxb, smax);
    scatter_k<<<NROW / TB, TB, 0, stream>>>(wfb, smax, v_buf, pos_enc, idxb, denom, num);
    fea_k<<<PLN / TB, TB, 0, stream>>>(num, denom, fp_a);
    dwconv3x3<<<PLN / TB, TB, 0, stream>>>(fp_a, wconv_w1, fp_b);
    dwconv3x3<<<PLN / TB, TB, 0, stream>>>(fp_b, wconv_w2, fp_a);
    res_conv3x3_wmma<<<dim3(8, 16, BB), TB, 0, stream>>>(fp_a, x_pl, c1x1_w, c1x1_b, fp_b);
    conv1x1_wmma<<<BB * 128, TB, 0, stream>>>(fp_b, convf_w, convf_b, fp_a);
    chan_stats<<<dim3(256, 32), TB, 0, stream>>>(fp_a, stats + 80, 32);
    bn_apply<<<PLN / TB, TB, 0, stream>>>(fp_a, stats + 80, convf_bn_g, convf_bn_b,
                                          xout[pl], 1.f / 65536.f, 0);
    sample_add<<<NROW / TB, TB, 0, stream>>>(xout[pl], p, cout, IJ[pl][0], IJ[pl][1]);
  }
}
